// RelationNet_64_63196148793923
// MI455X (gfx1250) — matrix operand order repaired and checked by reading, not compiled, not run on hardware
//
#include <hip/hip_runtime.h>
#include <hip/hip_bf16.h>

typedef __bf16 bf16_t;
typedef __attribute__((ext_vector_type(16))) __bf16 v16bf;
typedef __attribute__((ext_vector_type(8)))  float  v8f;

__device__ __forceinline__ float lrelu(float x) { return x >= 0.f ? x : 0.2f * x; }

// ---- WMMA fragment loaders (ISA 7.12.2, 16-bit, wave32) ----
// Vectorized A loader for kStride==1: per-lane data = two contiguous 16B chunks.
// A 16x32 (MxK): lane row M=lane&15; k = 16h + 0..15 contiguous.
__device__ __forceinline__ v16bf ld_frag_a_vec(const bf16_t* base, int rowStride) {
  int lane = threadIdx.x & 31;
  int row  = lane & 15, half = lane >> 4;
  const bf16_t* p = base + row * rowStride + 16 * half;
  union { v16bf v; uint4 q[2]; } u;
  u.q[0] = *(const uint4*)(p);        // k = 16h + 0..7  -> elems 0..7  (VGPR0-3)
  u.q[1] = *(const uint4*)(p + 8);    // k = 16h + 8..15 -> elems 8..15 (VGPR4-7)
  return u.v;
}
// Vectorized B loader for kStride==1: B 32x16 (KxN): lane col N=lane&15;
// k = 16*half + 0..15 contiguous (32 bytes).
__device__ __forceinline__ v16bf ld_frag_b_vec(const bf16_t* base, int colStride) {
  int lane = threadIdx.x & 31;
  int col  = lane & 15, half = lane >> 4;
  const bf16_t* p = base + col * colStride + 16 * half;
  union { v16bf v; uint4 q[2]; } u;
  u.q[0] = *(const uint4*)(p);
  u.q[1] = *(const uint4*)(p + 8);
  return u.v;
}

// ---- 0. cast conv weights [3][64][576] fp32 -> bf16 (same layout) ----
__global__ void k_cast_w(const float* w, bf16_t* o, int n) {
  int i = blockIdx.x * blockDim.x + threadIdx.x;
  if (i < n) o[i] = (bf16_t)w[i];
}

// ---- 1. conv1 (3->64, 3x3 pad1) + BN + LReLU + 2x2 maxpool -> [75,64,42,42] ----
__global__ void k_conv1(const float* __restrict__ in1, const float* __restrict__ in2,
                        const float* __restrict__ w, const float* __restrict__ g,
                        const float* __restrict__ bt, const float* __restrict__ mn,
                        const float* __restrict__ vr, float* __restrict__ out) {
  int tid = blockIdx.x * blockDim.x + threadIdx.x;
  const int total = 75 * 64 * 42 * 42;
  if (tid >= total) return;
  int ox = tid % 42; int t = tid / 42;
  int oy = t % 42;   t /= 42;
  int ch = t % 64;   int img = t / 64;
  const float* src = (img < 50) ? (in1 + (size_t)img * 3 * 84 * 84)
                                : (in2 + (size_t)(img - 50) * 3 * 84 * 84);
  float scale = g[ch] * rsqrtf(vr[ch] + 1e-5f);
  float shift = bt[ch] - mn[ch] * scale;
  float m = -3.4e38f;
#pragma unroll
  for (int py = 0; py < 2; ++py)
#pragma unroll
    for (int px = 0; px < 2; ++px) {
      int y = oy * 2 + py, x = ox * 2 + px;
      float acc = 0.f;
#pragma unroll
      for (int c = 0; c < 3; ++c)
#pragma unroll
        for (int dy = -1; dy <= 1; ++dy)
#pragma unroll
          for (int dx = -1; dx <= 1; ++dx) {
            int iy = y + dy, ix = x + dx;
            if (iy >= 0 && iy < 84 && ix >= 0 && ix < 84)
              acc += src[(c * 84 + iy) * 84 + ix] * w[((ch * 3 + c) * 3 + (dy + 1)) * 3 + (dx + 1)];
          }
      m = fmaxf(m, lrelu(acc * scale + shift));
    }
  out[tid] = m;
}

// ---- 2. conv 64->64 3x3 pad1 as implicit-GEMM WMMA + BN + LReLU ----
// grid(ntiles, nimg), block 128 (4 waves; wave w owns Cout rows 16w..16w+15)
// Templated on H/W so p/W, p%W are strength-reduced (no runtime div sequence).
template <int H, int Wd>
__global__ void k_conv_wmma(const float* __restrict__ in, const bf16_t* __restrict__ W,
                            const float* __restrict__ g, const float* __restrict__ bt,
                            const float* __restrict__ mn, const float* __restrict__ vr,
                            float* __restrict__ out) {
  constexpr int HW = H * Wd;
  int img = blockIdx.y;
  int nbase = blockIdx.x * 16;
  int wave = threadIdx.x >> 5;
  int lane = threadIdx.x & 31;
  int col = lane & 15, half = lane >> 4;
  int mbase = wave * 16;
  const float* src = in + (size_t)img * 64 * HW;
  int p = nbase + col;
  int oy = p / Wd, ox = p % Wd;
  bool pv = p < HW;
  v8f acc = {0.f, 0.f, 0.f, 0.f, 0.f, 0.f, 0.f, 0.f};
#pragma unroll 1
  for (int kc = 0; kc < 18; ++kc) {
    int kbase = kc * 32;
    v16bf a = ld_frag_a_vec(W + mbase * 576 + kbase, 576);
    v16bf b;
#pragma unroll
    for (int v = 0; v < 8; ++v)
#pragma unroll
      for (int e = 0; e < 2; ++e) {
        int k = kbase + 2 * v + 16 * half + e;
        int cin = k / 9, tap = k % 9;
        int iy = oy + tap / 3 - 1, ix = ox + tap % 3 - 1;
        float val = (pv && iy >= 0 && iy < H && ix >= 0 && ix < Wd)
                        ? src[cin * HW + iy * Wd + ix] : 0.f;
        b[2 * v + e] = (bf16_t)val;
      }
    acc = __builtin_amdgcn_wmma_f32_16x16x32_bf16(false, a, false, b, (short)0, acc, false, false);
  }
  if (pv) {
#pragma unroll
    for (int r = 0; r < 8; ++r) {
      int ch = mbase + r + 8 * half;
      float scale = g[ch] * rsqrtf(vr[ch] + 1e-5f);
      float v = acc[r] * scale + (bt[ch] - mn[ch] * scale);
      out[((size_t)img * 64 + ch) * HW + p] = lrelu(v);
    }
  }
}

// ---- 3. 2x2 maxpool [75,64,42,42] -> [75,64,21,21] ----
__global__ void k_pool(const float* __restrict__ in, float* __restrict__ out) {
  int tid = blockIdx.x * blockDim.x + threadIdx.x;
  const int total = 75 * 64 * 21 * 21;
  if (tid >= total) return;
  int ox = tid % 21; int t = tid / 21;
  int oy = t % 21;   int ci = t / 21;  // ci = img*64+ch
  const float* s = in + ((size_t)ci * 42 + oy * 2) * 42 + ox * 2;
  out[tid] = fmaxf(fmaxf(s[0], s[1]), fmaxf(s[42], s[43]));
}

// ---- 4a. pack query feats -> Bq[50][448][64] bf16 + q2sq[50][448] ----
__global__ void k_pack_q(const float* __restrict__ X, bf16_t* __restrict__ Bq,
                         float* __restrict__ qsq) {
  int tid = blockIdx.x * blockDim.x + threadIdx.x;
  if (tid >= 50 * 448) return;
  int b = tid / 448, pos = tid % 448;
  float s = 0.f;
  for (int d = 0; d < 64; ++d) {
    float v = (pos < 441) ? X[((size_t)b * 64 + d) * 441 + pos] : 0.f;
    s += v * v;
    Bq[(size_t)tid * 64 + d] = (bf16_t)v;
  }
  qsq[tid] = s;
}

// ---- 4b. pack support feats -> Bs[5][2208][64] bf16 + ssq[5][2208] ----
__global__ void k_pack_s(const float* __restrict__ X, bf16_t* __restrict__ Bs,
                         float* __restrict__ ssq) {
  int tid = blockIdx.x * blockDim.x + threadIdx.x;
  if (tid >= 5 * 2208) return;
  int c = tid / 2208, m = tid % 2208;
  if (m < 2205) {
    int sh = m / 441, pos = m % 441;
    int img = 50 + c * 5 + sh;
    float s = 0.f;
    for (int d = 0; d < 64; ++d) {
      float v = X[((size_t)img * 64 + d) * 441 + pos];
      s += v * v;
      Bs[(size_t)tid * 64 + d] = (bf16_t)v;
    }
    ssq[tid] = s;
  } else {
    for (int d = 0; d < 64; ++d) Bs[(size_t)tid * 64 + d] = (bf16_t)0.f;
    ssq[tid] = 1e30f;  // exp(-sqrt(1e30)) == 0 -> padded m contributes nothing
  }
}

// ---- 5. fused distance GEMM + exp + mean/var + classifier partial ----
// grid(28, 5, 50), block 256 (8 waves). partial[(b*5+c)*28+qt]
__global__ void k_dist(const bf16_t* __restrict__ Bq, const bf16_t* __restrict__ Bs,
                       const float* __restrict__ qsq, const float* __restrict__ ssqv,
                       const float* __restrict__ tP, const float* __restrict__ clsw,
                       float* __restrict__ part) {
  int qt = blockIdx.x, c = blockIdx.y, b = blockIdx.z;
  int wave = threadIdx.x >> 5, lane = threadIdx.x & 31;
  int col = lane & 15, half = lane >> 4;
  int qbase = qt * 16;
  const bf16_t* qb = Bq + ((size_t)b * 448 + qbase) * 64;
  v16bf a0 = ld_frag_a_vec(qb, 64);       // K = d 0..31
  v16bf a1 = ld_frag_a_vec(qb + 32, 64);  // K = d 32..63
  float invT = 1.f / tP[0];
  float qs[8];
#pragma unroll
  for (int r = 0; r < 8; ++r) qs[r] = qsq[b * 448 + qbase + r + 8 * half];
  float s1[8] = {0.f, 0.f, 0.f, 0.f, 0.f, 0.f, 0.f, 0.f};
  float s2[8] = {0.f, 0.f, 0.f, 0.f, 0.f, 0.f, 0.f, 0.f};
#pragma unroll 1
  for (int mt = wave; mt < 138; mt += 8) {
    int mbase = mt * 16;
    const bf16_t* sb = Bs + ((size_t)c * 2208 + mbase) * 64;
    v16bf b0 = ld_frag_b_vec(sb, 64);
    v16bf b1 = ld_frag_b_vec(sb + 32, 64);
    v8f d = {0.f, 0.f, 0.f, 0.f, 0.f, 0.f, 0.f, 0.f};
    d = __builtin_amdgcn_wmma_f32_16x16x32_bf16(false, a0, false, b0, (short)0, d, false, false);
    d = __builtin_amdgcn_wmma_f32_16x16x32_bf16(false, a1, false, b1, (short)0, d, false, false);
    float ss = ssqv[c * 2208 + mbase + col];
#pragma unroll
    for (int r = 0; r < 8; ++r) {
      float dis = fmaxf(qs[r] + ss - 2.f * d[r], 0.f) + 1e-9f;
      float e = __expf(-__fsqrt_rn(dis) * invT);
      s1[r] += e;
      s2[r] += e * e;
    }
  }
  // reduce over N (16 lanes per half-group; xor<=8 stays inside each half)
#pragma unroll
  for (int off = 1; off < 16; off <<= 1)
#pragma unroll
    for (int r = 0; r < 8; ++r) {
      s1[r] += __shfl_xor(s1[r], off, 32);
      s2[r] += __shfl_xor(s2[r], off, 32);
    }
  __shared__ float L1[8][16];
  __shared__ float L2[8][16];
  if (col == 0) {
#pragma unroll
    for (int r = 0; r < 8; ++r) {
      int M = r + 8 * half;
      L1[wave][M] = s1[r];
      L2[wave][M] = s2[r];
    }
  }
  __syncthreads();
  if (threadIdx.x < 16) {
    int M = threadIdx.x;
    float S1 = 0.f, S2 = 0.f;
#pragma unroll
    for (int w = 0; w < 8; ++w) { S1 += L1[w][M]; S2 += L2[w][M]; }
    const float n = 2205.f;
    float mean = S1 / n;
    float var  = (S2 - S1 * S1 / n) / (n - 1.f);
    float sim  = mean * sqrtf(var + 1e-9f);
    float x    = lrelu(sim);
    int q = qbase + M;
    float contrib = (q < 441) ? x * clsw[q] : 0.f;
#pragma unroll
    for (int off = 1; off < 16; off <<= 1) contrib += __shfl_xor(contrib, off, 32);
    if (M == 0) part[((size_t)b * 5 + c) * 28 + qt] = contrib;
  }
}

// ---- 6. final reduce: out[b,c] = sum of 28 partials ----
__global__ void k_final(const float* __restrict__ part, float* __restrict__ out) {
  int i = blockIdx.x * blockDim.x + threadIdx.x;
  if (i >= 250) return;
  float s = 0.f;
#pragma unroll
  for (int t = 0; t < 28; ++t) s += part[i * 28 + t];
  out[i] = s;
}

extern "C" void kernel_launch(void* const* d_in, const int* in_sizes, int n_in,
                              void* d_out, int out_size, void* d_ws, size_t ws_size,
                              hipStream_t stream) {
  const float* input1 = (const float*)d_in[0];
  const float* input2 = (const float*)d_in[1];
  const float* tP     = (const float*)d_in[2];
  const float* conv1w = (const float*)d_in[6];
  const float* convw  = (const float*)d_in[7];
  const float* bn_g   = (const float*)d_in[8];
  const float* bn_b   = (const float*)d_in[9];
  const float* bn_m   = (const float*)d_in[10];
  const float* bn_v   = (const float*)d_in[11];
  const float* clsw   = (const float*)d_in[12];
  float* out = (float*)d_out;

  char* ws = (char*)d_ws;
  // region A (33.87 MB): X1, later XP + X3 ; region B (33.87 MB): XC, later X4
  const size_t OFF_A   = 0;
  const size_t OFF_X3  = 16934400;            // inside A, after XP(8.47MB)
  const size_t OFF_B   = 33868800;
  const size_t OFF_WBF = 67737600;            // 3*64*576 bf16
  const size_t OFF_BQ  = OFF_WBF + 221184;    // 50*448*64 bf16
  const size_t OFF_BS  = OFF_BQ + 2867200;    // 5*2208*64 bf16
  const size_t OFF_QSQ = OFF_BS + 1413120;    // 50*448 f32
  const size_t OFF_SSQ = OFF_QSQ + 89600;     // 5*2208 f32
  const size_t OFF_PRT = OFF_SSQ + 44160;     // 50*5*28 f32

  float*  X1  = (float*)(ws + OFF_A);
  float*  XC  = (float*)(ws + OFF_B);
  float*  XP  = (float*)(ws + OFF_A);
  float*  X3  = (float*)(ws + OFF_X3);
  float*  X4  = (float*)(ws + OFF_B);
  bf16_t* Wbf = (bf16_t*)(ws + OFF_WBF);
  bf16_t* Bq  = (bf16_t*)(ws + OFF_BQ);
  bf16_t* Bs  = (bf16_t*)(ws + OFF_BS);
  float*  qsq = (float*)(ws + OFF_QSQ);
  float*  ssq = (float*)(ws + OFF_SSQ);
  float*  prt = (float*)(ws + OFF_PRT);

  // 0. cast weights (layout already [l][m][k])
  k_cast_w<<<(3 * 64 * 576 + 255) / 256, 256, 0, stream>>>(convw, Wbf, 3 * 64 * 576);
  // 1. conv1 + bn0 + lrelu + pool
  k_conv1<<<(75 * 64 * 42 * 42 + 255) / 256, 256, 0, stream>>>(
      input1, input2, conv1w, bn_g + 0, bn_b + 0, bn_m + 0, bn_v + 0, X1);
  // 2. conv2 (WMMA) + bn1 + lrelu  [42x42], N=1764 -> 111 tiles
  k_conv_wmma<42, 42><<<dim3(111, 75), 128, 0, stream>>>(
      X1, Wbf + 0 * 64 * 576, bn_g + 64, bn_b + 64, bn_m + 64, bn_v + 64, XC);
  // 3. pool 42->21
  k_pool<<<(75 * 64 * 21 * 21 + 255) / 256, 256, 0, stream>>>(XC, XP);
  // 4. conv3 (WMMA) + bn2 + lrelu  [21x21], N=441 -> 28 tiles
  k_conv_wmma<21, 21><<<dim3(28, 75), 128, 0, stream>>>(
      XP, Wbf + 1 * 64 * 576, bn_g + 128, bn_b + 128, bn_m + 128, bn_v + 128, X3);
  // 5. conv4 (WMMA) + bn3 + lrelu -> features [75,64,441]
  k_conv_wmma<21, 21><<<dim3(28, 75), 128, 0, stream>>>(
      X3, Wbf + 2 * 64 * 576, bn_g + 192, bn_b + 192, bn_m + 192, bn_v + 192, X4);
  // 6. pack bf16 fragments + squared norms
  k_pack_q<<<(50 * 448 + 255) / 256, 256, 0, stream>>>(X4, Bq, qsq);
  k_pack_s<<<(5 * 2208 + 255) / 256, 256, 0, stream>>>(X4, Bs, ssq);
  // 7. fused distance + exp + mean/var + classifier partials
  k_dist<<<dim3(28, 5, 50), 256, 0, stream>>>(Bq, Bs, qsq, ssq, tP, clsw, prt);
  // 8. final reduce -> out[50,5]
  k_final<<<1, 256, 0, stream>>>(prt, out);
}
